// QuantumGenerator_77120432767365
// MI455X (gfx1250) — matrix operand order repaired and checked by reading, not compiled, not run on hardware
//
#include <hip/hip_runtime.h>
#include <hip/hip_bf16.h>

typedef __attribute__((ext_vector_type(4)))  _Float16 v4h;
typedef __attribute__((ext_vector_type(8)))  _Float16 v8h;
typedef __attribute__((ext_vector_type(16))) _Float16 v16h;
typedef __attribute__((ext_vector_type(8)))  float    v8f;

#define NQ 4
#define DEPTH 3

// ---------------------------------------------------------------------------
// Workspace layout (bytes)
//  h1   : B*128 f32   @ 0
//  h2   : B*256 f32   @ 32MB
//  w2h  : 256*128 f16 @ 96MB
//  w3h  : 256*256 f16 @ 96MB + 64KB
//  stats: 1536 f32    @ 96MB + 192KB
//    [0..127] sum1, [128..255] ss1, [256..511] sum2, [512..767] ss2,
//    [768..895] sc1, [896..1023] sh1, [1024..1279] sc2, [1280..1535] sh2
// ---------------------------------------------------------------------------

__global__ void zero_kernel(float* p, int n) {
    int i = blockIdx.x * blockDim.x + threadIdx.x;
    if (i < n) p[i] = 0.0f;
}

__global__ void wconv_kernel(const float* __restrict__ w2,
                             const float* __restrict__ w3,
                             _Float16* __restrict__ w2h,
                             _Float16* __restrict__ w3h) {
    int i = blockIdx.x * blockDim.x + threadIdx.x;
    if (i < 256 * 128) w2h[i] = (_Float16)w2[i];
    if (i < 256 * 256) w3h[i] = (_Float16)w3[i];
}

// ---------------------------------------------------------------------------
// quantum statevector sim (4 qubits) + fc1 (4->128), one thread per element
// qubit q <-> bit (3-q) of the flat 16-state index
// ---------------------------------------------------------------------------
__global__ __launch_bounds__(256)
void sim_fc1_kernel(const float* __restrict__ x,
                    const float* __restrict__ qp,
                    const float* __restrict__ w1,
                    const float* __restrict__ b1,
                    float* __restrict__ h1, int B) {
    __shared__ float sw[128 * 4];
    __shared__ float sb[128];
    __shared__ float sq[DEPTH * 2 * NQ];
    const int tid = threadIdx.x;
    if (tid < DEPTH * 2 * NQ) sq[tid] = qp[tid];
    for (int i = tid; i < 512; i += 256) sw[i] = w1[i];
    if (tid < 128) sb[tid] = b1[tid];
    __syncthreads();

    const int b = blockIdx.x * 256 + tid;
    if (b >= B) return;

    float4 xv = *(const float4*)(x + (size_t)4 * b);
    float xa[4] = {xv.x, xv.y, xv.z, xv.w};

    float sr[16], si[16];
#pragma unroll
    for (int i = 0; i < 16; ++i) { sr[i] = 0.0f; si[i] = 0.0f; }
    sr[0] = 1.0f;

#pragma unroll
    for (int d = 0; d < DEPTH; ++d) {
        // RY(data + param) on each qubit (real rotation)
#pragma unroll
        for (int q = 0; q < NQ; ++q) {
            float t = xa[q] + sq[d * 8 + q];
            float s, c;
            __sincosf(0.5f * t, &s, &c);
            const int m = 1 << (3 - q);
#pragma unroll
            for (int i = 0; i < 16; ++i) {
                if (i & m) continue;
                const int j = i | m;
                float r0 = sr[i], i0 = si[i], r1 = sr[j], i1 = si[j];
                sr[i] = c * r0 - s * r1;  si[i] = c * i0 - s * i1;
                sr[j] = s * r0 + c * r1;  si[j] = s * i0 + c * i1;
            }
        }
        // ring of CNOTs: control q -> target (q+1)%4
#pragma unroll
        for (int q = 0; q < NQ; ++q) {
            const int mc = 1 << (3 - q);
            const int mt = 1 << (3 - ((q + 1) & 3));
#pragma unroll
            for (int i = 0; i < 16; ++i) {
                if ((i & mc) && !(i & mt)) {
                    const int j = i | mt;
                    float tr = sr[i]; sr[i] = sr[j]; sr[j] = tr;
                    float ti = si[i]; si[i] = si[j]; si[j] = ti;
                }
            }
        }
        // RZ(param): a0 *= e^{-i t/2}, a1 *= e^{+i t/2}
#pragma unroll
        for (int q = 0; q < NQ; ++q) {
            float th = sq[d * 8 + 4 + q];
            float s, c;
            __sincosf(0.5f * th, &s, &c);
            const int m = 1 << (3 - q);
#pragma unroll
            for (int i = 0; i < 16; ++i) {
                float r = sr[i], im = si[i];
                if (i & m) { sr[i] = r * c - im * s; si[i] = im * c + r * s; }
                else       { sr[i] = r * c + im * s; si[i] = im * c - r * s; }
            }
        }
    }

    // <Z_q> expectation values
    float z[4] = {0.0f, 0.0f, 0.0f, 0.0f};
#pragma unroll
    for (int i = 0; i < 16; ++i) {
        float p = sr[i] * sr[i] + si[i] * si[i];
#pragma unroll
        for (int q = 0; q < NQ; ++q)
            z[q] += (i & (1 << (3 - q))) ? -p : p;
    }

    // fc1: h1[j] = z . w1[j,:] + b1[j]
    float* hp = h1 + (size_t)b * 128;
#pragma unroll 4
    for (int j4 = 0; j4 < 32; ++j4) {
        float4 o;
        float* op = (float*)&o;
#pragma unroll
        for (int u = 0; u < 4; ++u) {
            const int j = 4 * j4 + u;
            op[u] = sb[j] + z[0] * sw[4 * j] + z[1] * sw[4 * j + 1]
                          + z[2] * sw[4 * j + 2] + z[3] * sw[4 * j + 3];
        }
        *(float4*)(hp + 4 * j4) = o;
    }
}

// ---------------------------------------------------------------------------
// per-channel sum and sum-of-squares over the batch axis
// ---------------------------------------------------------------------------
template <int C>
__global__ __launch_bounds__(256)
void colsum_kernel(const float* __restrict__ h, int B,
                   float* __restrict__ gsum, float* __restrict__ gss) {
    constexpr int C4 = C / 4;
    constexpr int RPB = 256 / C4;
    __shared__ float ssum[C];
    __shared__ float sss[C];
    const int tid = threadIdx.x;
    for (int i = tid; i < C; i += 256) { ssum[i] = 0.0f; sss[i] = 0.0f; }
    __syncthreads();

    const int c4 = tid % C4;
    const int r0 = blockIdx.x * RPB + tid / C4;
    const int rs = gridDim.x * RPB;
    float4 s = {0, 0, 0, 0}, q = {0, 0, 0, 0};
    for (int r = r0; r < B; r += rs) {
        float4 v = *(const float4*)(h + (size_t)r * C + 4 * c4);
        s.x += v.x; s.y += v.y; s.z += v.z; s.w += v.w;
        q.x += v.x * v.x; q.y += v.y * v.y; q.z += v.z * v.z; q.w += v.w * v.w;
    }
    atomicAdd(&ssum[4 * c4 + 0], s.x); atomicAdd(&sss[4 * c4 + 0], q.x);
    atomicAdd(&ssum[4 * c4 + 1], s.y); atomicAdd(&sss[4 * c4 + 1], q.y);
    atomicAdd(&ssum[4 * c4 + 2], s.z); atomicAdd(&sss[4 * c4 + 2], q.z);
    atomicAdd(&ssum[4 * c4 + 3], s.w); atomicAdd(&sss[4 * c4 + 3], q.w);
    __syncthreads();
    for (int i = tid; i < C; i += 256) {
        unsafeAtomicAdd(&gsum[i], ssum[i]);
        unsafeAtomicAdd(&gss[i], sss[i]);
    }
}

// ---------------------------------------------------------------------------
// bn finalize: scale = g * rsqrt(var+eps), shift = b - mean*scale
// ---------------------------------------------------------------------------
__global__ void bnfin_kernel(const float* __restrict__ sum,
                             const float* __restrict__ ss,
                             const float* __restrict__ g,
                             const float* __restrict__ b,
                             float* __restrict__ sc, float* __restrict__ sh,
                             int C, float invB) {
    int i = blockIdx.x * blockDim.x + threadIdx.x;
    if (i < C) {
        float m = sum[i] * invB;
        float v = ss[i] * invB - m * m;
        float s = g[i] * rsqrtf(v + 1e-5f);
        sc[i] = s;
        sh[i] = b[i] - m * s;
    }
}

// ---------------------------------------------------------------------------
// WMMA GEMM: out[M x NTOT] = act(BN(hin)) @ Wf16^T + bias  (opt. tanh)
//   hin: M x KDIM f32, BN+ReLU applied while staging to LDS as f16.
//   wh : NTOT x KDIM f16 (row major) -> B fragment is two 8-half K runs (8*hi, 16+8*hi).
//   256 threads = 8 waves; wave handles 16 rows x 64 cols (4 accum tiles).
//   v_wmma_f32_16x16x32_f16, K-loop = KDIM/32. All tile/K offsets fold into
//   load/store immediate offsets off a single per-wave base pointer.
// ---------------------------------------------------------------------------
template <int KDIM, int BM, int BN, bool TANH>
__global__ __launch_bounds__(256)
void gemm_bn_wmma_kernel(const float* __restrict__ hin,
                         const float* __restrict__ scale,
                         const float* __restrict__ shift,
                         const _Float16* __restrict__ wh,
                         const float* __restrict__ bias,
                         float* __restrict__ out, int NTOT) {
    constexpr int STR = KDIM + 8;             // padded LDS row stride (halves)
    __shared__ _Float16 smem[BM * STR];
    const int tid = threadIdx.x;
    const int rowbase = blockIdx.x * BM;
    const int colbase = blockIdx.y * BN;

    // ---- prefetch this block's weight slice (overlaps with A staging) ----
    {
        const char* wbase = (const char*)(wh + (size_t)colbase * KDIM);
        constexpr int WBYTES = BN * KDIM * 2;
        for (int off = tid * 128; off < WBYTES; off += 256 * 128)
            __builtin_prefetch(wbase + off, 0, 3);
    }

    // ---- stage A block: f32 -> BN -> ReLU -> f16 in LDS ----
    constexpr int K4 = KDIM / 4;
    constexpr int TOT4 = BM * K4;             // 4096 float4s for both configs
#pragma unroll
    for (int i = 0; i < TOT4 / 256; ++i) {
        const int idx = i * 256 + tid;
        const int row = idx / K4;
        const int c4 = idx % K4;
        const float4 v = *(const float4*)(hin + (size_t)(rowbase + row) * KDIM + 4 * c4);
        const float4 sc = *(const float4*)(scale + 4 * c4);
        const float4 sf = *(const float4*)(shift + 4 * c4);
        v4h h;
        h[0] = (_Float16)fmaxf(v.x * sc.x + sf.x, 0.0f);
        h[1] = (_Float16)fmaxf(v.y * sc.y + sf.y, 0.0f);
        h[2] = (_Float16)fmaxf(v.z * sc.z + sf.z, 0.0f);
        h[3] = (_Float16)fmaxf(v.w * sc.w + sf.w, 0.0f);
        *(v4h*)(&smem[row * STR + 4 * c4]) = h;
    }
    __syncthreads();

    // ---- wave tiling ----
    const int wave = tid >> 5;
    const int lane = tid & 31;
    const int lrow = lane & 15;
    const int hi = lane >> 4;
    constexpr int RS = BM / 16;               // row strips per block
    const int wrow = (wave % RS) * 16;        // wave's row strip within block
    const int wcol = (wave / RS) * 64;        // wave's 64-col group within block

    // per-wave base pointers; all tile/K offsets below are compile-time
    const _Float16* ap = &smem[(wrow + lrow) * STR + 8 * hi];
    const _Float16* bp = wh + (size_t)(colbase + wcol + lrow) * KDIM + 8 * hi;

    v8f acc[4] = {};

#pragma unroll
    for (int ks = 0; ks < KDIM / 32; ++ks) {
        const int k0 = ks * 32;
        // A fragment (16x32 f16): lane = row (lrow), half-lane split on K
        const v8h alo = *(const v8h*)(ap + k0);       // K = k0 + 8*hi + [0..7]
        const v8h ahi = *(const v8h*)(ap + k0 + 16);  // K = k0 + 16 + 8*hi + [0..7]
        const v16h a = __builtin_shufflevector(alo, ahi, 0, 1, 2, 3, 4, 5, 6, 7,
                                               8, 9, 10, 11, 12, 13, 14, 15);
#pragma unroll
        for (int t = 0; t < 4; ++t) {
            // B fragment (32x16 f16): lane = column N, same K split as A
            const _Float16* bt = bp + t * 16 * KDIM + k0;
            const v8h blo = *(const v8h*)bt;
            const v8h bhi = *(const v8h*)(bt + 16);
            const v16h bfr = __builtin_shufflevector(blo, bhi, 0, 1, 2, 3, 4, 5, 6, 7,
                                                     8, 9, 10, 11, 12, 13, 14, 15);
            acc[t] = __builtin_amdgcn_wmma_f32_16x16x32_f16(
                false, a, false, bfr, (short)0, acc[t], false, false);
        }
    }

    // ---- epilogue: bias (+ tanh); C/D layout: VGPR r -> M = r + 8*hi ----
    float* op = out + (size_t)(rowbase + wrow + 8 * hi) * NTOT + colbase + wcol + lrow;
#pragma unroll
    for (int t = 0; t < 4; ++t) {
        const float bn = bias[colbase + wcol + t * 16 + lrow];
#pragma unroll
        for (int r = 0; r < 8; ++r) {
            float v = acc[t][r] + bn;
            if (TANH) {
                // final output: never re-read on device -> non-temporal store
                __builtin_nontemporal_store(tanhf(v), op + t * 16 + r * NTOT);
            } else {
                op[t * 16 + r * NTOT] = v;
            }
        }
    }
}

// ---------------------------------------------------------------------------
// host launcher
// ---------------------------------------------------------------------------
extern "C" void kernel_launch(void* const* d_in, const int* in_sizes, int n_in,
                              void* d_out, int out_size, void* d_ws, size_t ws_size,
                              hipStream_t stream) {
    const float* x      = (const float*)d_in[0];
    const float* qp     = (const float*)d_in[1];
    const float* fc1_w  = (const float*)d_in[2];
    const float* fc1_b  = (const float*)d_in[3];
    const float* bn1_g  = (const float*)d_in[4];
    const float* bn1_b  = (const float*)d_in[5];
    const float* fc2_w  = (const float*)d_in[6];
    const float* fc2_b  = (const float*)d_in[7];
    const float* bn2_g  = (const float*)d_in[8];
    const float* bn2_b  = (const float*)d_in[9];
    const float* fc3_w  = (const float*)d_in[10];
    const float* fc3_b  = (const float*)d_in[11];
    float* out = (float*)d_out;

    const int B = in_sizes[0] / 4;

    char* ws = (char*)d_ws;
    float*    h1    = (float*)(ws);
    float*    h2    = (float*)(ws + (size_t)B * 128 * 4);
    _Float16* w2h   = (_Float16*)(ws + (size_t)B * 128 * 4 + (size_t)B * 256 * 4);
    _Float16* w3h   = (_Float16*)((char*)w2h + 256 * 128 * 2);
    float*    stats = (float*)((char*)w3h + 256 * 256 * 2);
    float* sum1 = stats;         float* ss1 = stats + 128;
    float* sum2 = stats + 256;   float* ss2 = stats + 512;
    float* sc1  = stats + 768;   float* sh1 = stats + 896;
    float* sc2  = stats + 1024;  float* sh2 = stats + 1280;

    // 1) zero BN accumulators
    zero_kernel<<<3, 256, 0, stream>>>(stats, 768);
    // 2) weights -> f16
    wconv_kernel<<<256, 256, 0, stream>>>(fc2_w, fc3_w, w2h, w3h);
    // 3) quantum sim + fc1
    sim_fc1_kernel<<<B / 256, 256, 0, stream>>>(x, qp, fc1_w, fc1_b, h1, B);
    // 4) bn1 stats + finalize
    colsum_kernel<128><<<1024, 256, 0, stream>>>(h1, B, sum1, ss1);
    bnfin_kernel<<<1, 128, 0, stream>>>(sum1, ss1, bn1_g, bn1_b, sc1, sh1, 128, 1.0f / B);
    // 5) fc2 via WMMA: (B x 128) x (128 x 256)
    gemm_bn_wmma_kernel<128, 128, 64, false>
        <<<dim3(B / 128, 256 / 64), 256, 0, stream>>>(h1, sc1, sh1, w2h, fc2_b, h2, 256);
    // 6) bn2 stats + finalize
    colsum_kernel<256><<<1024, 256, 0, stream>>>(h2, B, sum2, ss2);
    bnfin_kernel<<<1, 256, 0, stream>>>(sum2, ss2, bn2_g, bn2_b, sc2, sh2, 256, 1.0f / B);
    // 7) fc3 via WMMA + tanh: (B x 256) x (256 x 256) -> out
    gemm_bn_wmma_kernel<256, 64, 128, true>
        <<<dim3(B / 64, 256 / 128), 256, 0, stream>>>(h2, sc2, sh2, w3h, fc3_b, out, 256);
}
